// tDKFN_42588895707896
// MI455X (gfx1250) — matrix operand order repaired and checked by reading, not compiled, not run on hardware
//
#include <hip/hip_runtime.h>

typedef __bf16 bf16_t;
typedef __attribute__((ext_vector_type(16))) __bf16 v16bf;
typedef __attribute__((ext_vector_type(8)))  __bf16 v8bf;
typedef __attribute__((ext_vector_type(8)))  float  v8f;

#define NF   1024
#define BB   2048
#define GIN  5120          // K*N + 2N
#define RIN  3072          // N + 2N
#define HN   (BB * NF)     // 2,097,152 elements per [B,N] tensor

// ---------- helpers ----------
__device__ __forceinline__ bf16_t f2bf(float f) {
  union { float f; unsigned u; } x; x.f = f;
  unsigned u = x.u;
  unsigned r = u + 0x7FFFu + ((u >> 16) & 1u);   // round-to-nearest-even
  unsigned short h = (unsigned short)(r >> 16);
  bf16_t o; __builtin_memcpy(&o, &h, sizeof(o));
  return o;
}

// ---------- conversion / assembly kernels ----------
__global__ void k_f32_to_bf16(const float* __restrict__ s, bf16_t* __restrict__ d, int n) {
  int i = blockIdx.x * blockDim.x + threadIdx.x;
  if (i < n) d[i] = f2bf(s[i]);
}

__global__ void k_mask_weights(const float* __restrict__ A, const float* __restrict__ W,
                               bf16_t* __restrict__ d, int n) {
  int i = blockIdx.x * blockDim.x + threadIdx.x;
  if (i < n) d[i] = f2bf(A[i] * W[i]);
}

__global__ void k_fill_combined_tail(const float* __restrict__ Hs, const float* __restrict__ pred,
                                     bf16_t* __restrict__ comb) {
  int i = blockIdx.x * blockDim.x + threadIdx.x;     // over B*N
  if (i >= HN) return;
  int b = i >> 10, m = i & 1023;
  comb[(size_t)b * GIN + 3 * NF + m] = f2bf(Hs[i]);
  comb[(size_t)b * GIN + 4 * NF + m] = f2bf(pred[i]);
}

__global__ void k_fill_rcombined(const float* __restrict__ it, const float* __restrict__ rh,
                                 const float* __restrict__ pred, bf16_t* __restrict__ rc) {
  int i = blockIdx.x * blockDim.x + threadIdx.x;     // over B*N
  if (i >= HN) return;
  int b = i >> 10, m = i & 1023;
  rc[(size_t)b * RIN + m]          = f2bf(it[i]);
  rc[(size_t)b * RIN + NF + m]     = f2bf(rh[i]);
  rc[(size_t)b * RIN + 2 * NF + m] = f2bf(pred[i]);
}

// nbr[m] = A_list[-1][m,:] . Neighbor_weight
__global__ void k_nbr(const float* __restrict__ Alast, const float* __restrict__ Nw,
                      float* __restrict__ nbr) {
  int m = blockIdx.x * blockDim.x + threadIdx.x;
  if (m >= NF) return;
  float s = 0.f;
  for (int n = 0; n < NF; ++n) s += Alast[(size_t)m * NF + n] * Nw[n];
  nbr[m] = s;
}

// ---------- WMMA bf16 GEMM: C[M x Nout] = act(A[M x K] @ W[Nout x K]^T + bias) ----------
// 8 waves / block arranged 2x4; each wave owns a 32x32 register tile
// (4 accumulators, 2 A frags x 2 B frags per K-step -> 2 loads per WMMA).
// Block covers 64 rows x 128 cols of C.
__global__ void __launch_bounds__(256) k_gemm_wmma(
    const bf16_t* __restrict__ A, int lda,
    const bf16_t* __restrict__ W, int ldw,
    const float*  __restrict__ bias,
    float*        __restrict__ C, int ldc,
    bf16_t*       __restrict__ Cbf, int ldcb,
    int Kdim, int act)
{
  const int lane  = threadIdx.x & 31;
  const int wave  = threadIdx.x >> 5;
  const int tileN = blockIdx.x * 128 + (wave & 3) * 32;
  const int tileM = blockIdx.y * 64  + (wave >> 2) * 32;
  const int l15   = lane & 15;
  const int hi    = lane >> 4;

  // A frag (16x32 bf16, ISA 7.12.2): lane L<16 row=L, K {kb..kb+7},{16+kb..}; kb = 8*(L>=16)
  const bf16_t* ap0 = A + (size_t)(tileM + l15) * lda + hi * 8;
  const bf16_t* ap1 = ap0 + (size_t)16 * lda;
  // B frag (32x16): lane L holds col (L&15), K {kb..kb+7},{16+kb..}; kb = 8*(L>=16)
  const bf16_t* bp0 = W + (size_t)(tileN + l15) * ldw + hi * 8;
  const bf16_t* bp1 = bp0 + (size_t)16 * ldw;

  v8f acc00 = 0.0f, acc01 = 0.0f, acc10 = 0.0f, acc11 = 0.0f;
  for (int kk = 0; kk < Kdim; kk += 32) {
    union { v16bf v; v8bf h[2]; } a0, a1, b0, b1;
    a0.h[0] = *(const v8bf*)(ap0);       a0.h[1] = *(const v8bf*)(ap0 + 16);
    a1.h[0] = *(const v8bf*)(ap1);       a1.h[1] = *(const v8bf*)(ap1 + 16);
    b0.h[0] = *(const v8bf*)(bp0);       b0.h[1] = *(const v8bf*)(bp0 + 16);
    b1.h[0] = *(const v8bf*)(bp1);       b1.h[1] = *(const v8bf*)(bp1 + 16);
    __builtin_prefetch(ap0 + 64, 0, 0);
    __builtin_prefetch(ap1 + 64, 0, 0);
    __builtin_prefetch(bp0 + 64, 0, 0);
    __builtin_prefetch(bp1 + 64, 0, 0);
    acc00 = __builtin_amdgcn_wmma_f32_16x16x32_bf16(false, a0.v, false, b0.v,
                                                    (short)0, acc00, false, false);
    acc01 = __builtin_amdgcn_wmma_f32_16x16x32_bf16(false, a0.v, false, b1.v,
                                                    (short)0, acc01, false, false);
    acc10 = __builtin_amdgcn_wmma_f32_16x16x32_bf16(false, a1.v, false, b0.v,
                                                    (short)0, acc10, false, false);
    acc11 = __builtin_amdgcn_wmma_f32_16x16x32_bf16(false, a1.v, false, b1.v,
                                                    (short)0, acc11, false, false);
    ap0 += 32; ap1 += 32; bp0 += 32; bp1 += 32;
  }

  // C/D layout: VGPR r -> row (r + 8*hi), col = lane&15
  v8f accs[4] = {acc00, acc01, acc10, acc11};
#pragma unroll
  for (int t = 0; t < 4; ++t) {
    const int rowBase = tileM + ((t >> 1) ? 16 : 0) + hi * 8;
    const int col     = tileN + ((t & 1)  ? 16 : 0) + l15;
    const float bz = bias ? bias[col] : 0.0f;
#pragma unroll
    for (int r = 0; r < 8; ++r) {
      int row = rowBase + r;
      float v = accs[t][r] + bz;
      if (act == 1)      v = 1.0f / (1.0f + __expf(-v));   // sigmoid
      else if (act == 2) v = tanhf(v);                     // tanh
      C[(size_t)row * ldc + col] = v;
      if (Cbf) Cbf[(size_t)row * ldcb + col] = f2bf(v);
    }
  }
}

// ---------- reductions for unbiased variance ----------
__global__ void k_reduce_partial(const float* __restrict__ x, int n,
                                 float* __restrict__ ps, float* __restrict__ pq) {
  __shared__ float s1[256], s2[256];
  float a = 0.f, b = 0.f;
  for (int i = blockIdx.x * blockDim.x + threadIdx.x; i < n; i += gridDim.x * blockDim.x) {
    float v = x[i]; a += v; b += v * v;
  }
  s1[threadIdx.x] = a; s2[threadIdx.x] = b;
  __syncthreads();
  for (int s = 128; s > 0; s >>= 1) {
    if ((int)threadIdx.x < s) {
      s1[threadIdx.x] += s1[threadIdx.x + s];
      s2[threadIdx.x] += s2[threadIdx.x + s];
    }
    __syncthreads();
  }
  if (threadIdx.x == 0) { ps[blockIdx.x] = s1[0]; pq[blockIdx.x] = s2[0]; }
}

__global__ void k_reduce_final(const float* __restrict__ ps, const float* __restrict__ pq,
                               int nb, float n, float* __restrict__ var) {
  __shared__ float s1[256], s2[256];
  float a = 0.f, b = 0.f;
  for (int i = threadIdx.x; i < nb; i += 256) { a += ps[i]; b += pq[i]; }
  s1[threadIdx.x] = a; s2[threadIdx.x] = b;
  __syncthreads();
  for (int s = 128; s > 0; s >>= 1) {
    if ((int)threadIdx.x < s) {
      s1[threadIdx.x] += s1[threadIdx.x + s];
      s2[threadIdx.x] += s2[threadIdx.x + s];
    }
    __syncthreads();
  }
  if (threadIdx.x == 0) var[0] = (s2[0] - s1[0] * s1[0] / n) / (n - 1.0f);
}

// ---------- elementwise epilogues ----------
__global__ void k_cell_hidden(const float* __restrict__ gF, const float* __restrict__ gI,
                              const float* __restrict__ gO, const float* __restrict__ gC,
                              const float* __restrict__ CellState, const float* __restrict__ nbr,
                              float* __restrict__ CellOut, float* __restrict__ HiddenBuf) {
  int i = blockIdx.x * blockDim.x + threadIdx.x;
  if (i >= HN) return;
  int m = i & 1023;
  float cell = gF[i] * (CellState[i] * nbr[m]) + gI[i] * gC[i];
  CellOut[i] = cell;
  HiddenBuf[i] = gO[i] * tanhf(cell);
}

__global__ void k_rcell_rhidden(const float* __restrict__ gF, const float* __restrict__ gI,
                                const float* __restrict__ gO, const float* __restrict__ gC,
                                const float* __restrict__ rCellState,
                                float* __restrict__ rCellOut, float* __restrict__ rHiddenBuf) {
  int i = blockIdx.x * blockDim.x + threadIdx.x;
  if (i >= HN) return;
  float cell = gF[i] * rCellState[i] + gI[i] * gC[i];
  rCellOut[i] = cell;
  rHiddenBuf[i] = gO[i] * tanhf(cell);
}

__global__ void k_final(const float* __restrict__ Hid, const float* __restrict__ rHid,
                        const float* __restrict__ v1p, const float* __restrict__ v2p,
                        const float* __restrict__ cp, float* __restrict__ out) {
  int i = blockIdx.x * blockDim.x + threadIdx.x;
  if (i >= HN) return;
  float v1 = v1p[0], v2 = v2p[0], c = cp[0];
  float p = (Hid[i] * v1 * c + rHid[i] * v2) / (v1 + v2 * c);
  out[(size_t)7 * HN + i] = p;            // pred_new
  out[i]                  = Hid[i] - p;   // Hidden_out
  out[(size_t)5 * HN + i] = rHid[i] - p;  // rHidden_out
}

// ---------- host-side orchestration ----------
extern "C" void kernel_launch(void* const* d_in, const int* in_sizes, int n_in,
                              void* d_out, int out_size, void* d_ws, size_t ws_size,
                              hipStream_t stream) {
  (void)in_sizes; (void)n_in; (void)out_size; (void)ws_size;
  const float* input   = (const float*)d_in[0];
  const float* input_t = (const float*)d_in[1];
  const float* Hs      = (const float*)d_in[2];
  const float* Cs      = (const float*)d_in[3];
  const float* rHs     = (const float*)d_in[4];
  const float* rCs     = (const float*)d_in[5];
  const float* pred    = (const float*)d_in[6];
  const float* A_list  = (const float*)d_in[7];
  const float* gc_W    = (const float*)d_in[8];
  const float* gW[4]   = {(const float*)d_in[9],  (const float*)d_in[11],
                          (const float*)d_in[13], (const float*)d_in[15]};
  const float* gB[4]   = {(const float*)d_in[10], (const float*)d_in[12],
                          (const float*)d_in[14], (const float*)d_in[16]};
  const float* Nw      = (const float*)d_in[17];
  const float* rW[4]   = {(const float*)d_in[18], (const float*)d_in[20],
                          (const float*)d_in[22], (const float*)d_in[24]};
  const float* rB[4]   = {(const float*)d_in[19], (const float*)d_in[21],
                          (const float*)d_in[23], (const float*)d_in[25]};
  const float* cp      = (const float*)d_in[26];
  float* out = (float*)d_out;

  // workspace carve-out (256B aligned)
  char* w = (char*)d_ws;
  auto carve = [&](size_t bytes) { char* p = w; w += (bytes + 255) & ~(size_t)255; return p; };
  bf16_t* Xbf   = (bf16_t*)carve((size_t)HN * 2);                 // input bf16
  bf16_t* MWbf  = (bf16_t*)carve((size_t)3 * NF * NF * 2);        // masked gc weights
  bf16_t* comb  = (bf16_t*)carve((size_t)BB * GIN * 2);           // combined bf16
  bf16_t* rcomb = (bf16_t*)carve((size_t)BB * RIN * 2);           // rcombined bf16
  bf16_t* wbuf  = (bf16_t*)carve((size_t)NF * GIN * 2);           // reused weight staging
  float*  gates = (float*)carve((size_t)4 * HN * 4);              // f,i,o,C (reused for r)
  float*  Hbuf  = (float*)carve((size_t)HN * 4);
  float*  rHbuf = (float*)carve((size_t)HN * 4);
  float*  nbr   = (float*)carve((size_t)NF * 4);
  float*  ps    = (float*)carve(1024 * 4);
  float*  pq    = (float*)carve(1024 * 4);
  float*  var1  = (float*)carve(256);
  float*  var2  = (float*)carve(256);

  const int T = 256;
  const int gHN = (HN + T - 1) / T;

  // 1) operand conversion / assembly
  k_f32_to_bf16<<<gHN, T, 0, stream>>>(input, Xbf, HN);
  {
    int n = 3 * NF * NF;
    k_mask_weights<<<(n + T - 1) / T, T, 0, stream>>>(A_list, gc_W, MWbf, n);
  }

  // 2) gc GEMM: [2048x1024] @ [3072x1024]^T -> fp32 gc (output) + bf16 into combined[:, :3072]
  k_gemm_wmma<<<dim3(3 * NF / 128, BB / 64), T, 0, stream>>>(
      Xbf, NF, MWbf, NF, nullptr, out + (size_t)2 * HN, 3 * NF, comb, GIN, NF, 0);

  // 3) combined tail + neighbor vector
  k_fill_combined_tail<<<gHN, T, 0, stream>>>(Hs, pred, comb);
  k_nbr<<<NF / T, T, 0, stream>>>(A_list + (size_t)2 * NF * NF, Nw, nbr);

  // 4) four gate GEMMs (sigmoid x3, tanh x1), weights staged bf16 through wbuf
  for (int g = 0; g < 4; ++g) {
    int n = NF * GIN;
    k_f32_to_bf16<<<(n + T - 1) / T, T, 0, stream>>>(gW[g], wbuf, n);
    k_gemm_wmma<<<dim3(NF / 128, BB / 64), T, 0, stream>>>(
        comb, GIN, wbuf, GIN, gB[g], gates + (size_t)g * HN, NF, nullptr, 0,
        GIN, g < 3 ? 1 : 2);
  }

  // 5) Cell / Hidden
  k_cell_hidden<<<gHN, T, 0, stream>>>(gates, gates + (size_t)HN, gates + (size_t)2 * HN,
                                       gates + (size_t)3 * HN, Cs, nbr,
                                       out + (size_t)HN, Hbuf);

  // 6) r-branch GEMMs (reuse gates buffer)
  k_fill_rcombined<<<gHN, T, 0, stream>>>(input_t, rHs, pred, rcomb);
  for (int g = 0; g < 4; ++g) {
    int n = NF * RIN;
    k_f32_to_bf16<<<(n + T - 1) / T, T, 0, stream>>>(rW[g], wbuf, n);
    k_gemm_wmma<<<dim3(NF / 128, BB / 64), T, 0, stream>>>(
        rcomb, RIN, wbuf, RIN, rB[g], gates + (size_t)g * HN, NF, nullptr, 0,
        RIN, g < 3 ? 1 : 2);
  }
  k_rcell_rhidden<<<gHN, T, 0, stream>>>(gates, gates + (size_t)HN, gates + (size_t)2 * HN,
                                         gates + (size_t)3 * HN, rCs,
                                         out + (size_t)6 * HN, rHbuf);

  // 7) variances (ddof=1): var1 over input_t, var2 over fp32 gc in d_out
  k_reduce_partial<<<1024, T, 0, stream>>>(input_t, HN, ps, pq);
  k_reduce_final<<<1, T, 0, stream>>>(ps, pq, 1024, (float)HN, var1);
  k_reduce_partial<<<1024, T, 0, stream>>>(out + (size_t)2 * HN, 3 * HN, ps, pq);
  k_reduce_final<<<1, T, 0, stream>>>(ps, pq, 1024, (float)(3 * HN), var2);

  // 8) pred_new blend + Hidden_out / rHidden_out
  k_final<<<gHN, T, 0, stream>>>(Hbuf, rHbuf, var1, var2, cp, out);
}
